// FC_conv_55095840473448
// MI455X (gfx1250) — matrix operand order repaired and checked by reading, not compiled, not run on hardware
//
#include <hip/hip_runtime.h>

// Problem constants
#define BB 32
#define RR 256
#define TT 512
#define CC 16
#define C1 64
#define C2 128

typedef __attribute__((ext_vector_type(16))) _Float16 v16h;
typedef __attribute__((ext_vector_type(8)))  _Float16 v8h;
typedef __attribute__((ext_vector_type(8)))  float    v8f;

__device__ __forceinline__ v8f wmma_f16(v16h a, v16h b, v8f c) {
    // D = A(16x32 f16) * B(32x16 f16) + C(16x16 f32)
    return __builtin_amdgcn_wmma_f32_16x16x32_f16(
        /*neg_a=*/false, a, /*neg_b=*/false, b,
        /*c_mod=*/(short)0, c, /*reuse_a=*/false, /*reuse_b=*/false);
}

// ---------------------------------------------------------------------------
// Kernel 1: center x over T per (b,c,r), emit f16 xc[B,C,R,T] and rnorm[B,C,R].
// Block = 256 threads handles one (b,r): loads x[b,r,:,:] fully coalesced.
// thread -> (c = tid&15, tb = tid>>4), t = tb + 16*j  => flat = tid + 256*j.
// ---------------------------------------------------------------------------
__global__ __launch_bounds__(256) void prep_xc(const float* __restrict__ x,
                                               _Float16* __restrict__ xc,
                                               float* __restrict__ rnorm) {
    const int tid = threadIdx.x;
    const int b = blockIdx.x >> 8;    // / RR
    const int r = blockIdx.x & 255;
    const int cc = tid & 15;
    const int tb = tid >> 4;

    __shared__ float red[256];
    __shared__ _Float16 xch[16][520];   // padded rows (1040B stride, 16B aligned)

    const float* xp = x + (size_t)(b * RR + r) * TT * CC;
    float v[32];
    float s = 0.f;
#pragma unroll
    for (int j = 0; j < 32; ++j) { v[j] = xp[tid + 256 * j]; s += v[j]; }

    red[tid] = s; __syncthreads();
    if (tid < 128) red[tid] += red[tid + 128]; __syncthreads();
    if (tid < 64)  red[tid] += red[tid + 64];  __syncthreads();
    if (tid < 32)  red[tid] += red[tid + 32];  __syncthreads();
    if (tid < 16)  red[tid] += red[tid + 16];  __syncthreads();
    const float mean = red[cc] * (1.f / 512.f);
    __syncthreads();

    float ss = 0.f;
#pragma unroll
    for (int j = 0; j < 32; ++j) {
        float d = v[j] - mean;
        ss += d * d;
        xch[cc][tb + 16 * j] = (_Float16)d;
    }
    red[tid] = ss; __syncthreads();
    if (tid < 128) red[tid] += red[tid + 128]; __syncthreads();
    if (tid < 64)  red[tid] += red[tid + 64];  __syncthreads();
    if (tid < 32)  red[tid] += red[tid + 32];  __syncthreads();
    if (tid < 16)  red[tid] += red[tid + 16];  __syncthreads();
    if (tid < 16)
        rnorm[((size_t)b * CC + tid) * RR + r] = rsqrtf(red[tid] + 1e-20f);
    __syncthreads();

    // coalesced f16 writeout: thread handles channel oc, 32-half segment seg
    const int oc = tid >> 4, seg = tid & 15;
    _Float16* dst = xc + ((size_t)(b * CC + oc) * RR + r) * TT + seg * 32;
    const v8h* sp = (const v8h*)&xch[oc][seg * 32];
    v8h* dp = (v8h*)dst;
    dp[0] = sp[0]; dp[1] = sp[1]; dp[2] = sp[2]; dp[3] = sp[3];
}

// ---------------------------------------------------------------------------
// Kernel 2: W_edge fp32 -> f16 (layout preserved: [C1][C][R])
// ---------------------------------------------------------------------------
__global__ __launch_bounds__(256) void conv_wedge(const float* __restrict__ w,
                                                  _Float16* __restrict__ wh) {
    int i = blockIdx.x * 256 + threadIdx.x;
    if (i < C1 * CC * RR) wh[i] = (_Float16)w[i];
}

// ---------------------------------------------------------------------------
// Kernel 3: fused correlation GEMM + pcc scaling + EdgeConv contraction.
// One block per (b, c, row-tile of 16). 8 waves.
// Phase 1: cov strip 16x256 via v_wmma_f32_16x16x32_f16 (2 col-tiles/wave),
//          scaled by rnorm_r*rnorm_k into LDS.
// Phase 2: waves 0..3: pcc_strip(16x256) x W_edge[:,c,:]^T (64 cols) via WMMA,
//          atomic-accumulate into e[b, rows, :] over c.
// ---------------------------------------------------------------------------
__global__ __launch_bounds__(256) void corr_edge(const _Float16* __restrict__ xc,
                                                 const float* __restrict__ rnorm,
                                                 const _Float16* __restrict__ Wh,
                                                 float* __restrict__ e) {
    const int tid  = threadIdx.x;
    const int lane = tid & 31;
    const int wave = tid >> 5;
    const int lm   = lane & 15;   // A: row M / B: col N (ISA lane layout)
    const int hi   = lane >> 4;   // selects K sub-block per ISA layout

    const int bid = blockIdx.x;          // B*C*16 = 8192 blocks
    const int rt  = bid & 15;            // row tile
    const int c   = (bid >> 4) & 15;
    const int b   = bid >> 8;

    __shared__ float rn_s[256];
    __shared__ float pcc_s[16 * 260];    // 16 rows x 256 cols, padded stride

    const size_t bc = (size_t)(b * CC + c);
    const _Float16* xbc = xc + bc * RR * TT;
    rn_s[tid] = rnorm[bc * RR + tid];

    const _Float16* arow  = xbc + (size_t)(rt * 16 + lm) * TT;
    const int ct0 = wave * 2, ct1 = wave * 2 + 1;
    const _Float16* brow0 = xbc + (size_t)(ct0 * 16 + lm) * TT;
    const _Float16* brow1 = xbc + (size_t)(ct1 * 16 + lm) * TT;

    v8f acc0 = {}; v8f acc1 = {};
    for (int kb = 0; kb < TT; kb += 32) {
        // A 16x32 f16: lane row lm; 16 contiguous K at kb + hi*16
        v8h a0 = *(const v8h*)(arow + kb + hi * 16);
        v8h a1 = *(const v8h*)(arow + kb + hi * 16 + 8);
        v16h a = __builtin_shufflevector(a0, a1,
                 0,1,2,3,4,5,6,7, 8,9,10,11,12,13,14,15);
        // B 32x16 f16: lane col lm; 16 contiguous K at kb + hi*16
        v16h b0 = *(const v16h*)(brow0 + kb + hi * 16);
        v16h b1 = *(const v16h*)(brow1 + kb + hi * 16);
        acc0 = wmma_f16(a, b0, acc0);
        acc1 = wmma_f16(a, b1, acc1);
    }
    __syncthreads();   // rn_s visible, pcc_s free

    // scale cov -> pcc and park the 16x256 strip in LDS
#pragma unroll
    for (int v = 0; v < 8; ++v) {
        const int m = v + hi * 8;              // C/D layout: M = v + 8*(lane>=16)
        const float rr = rn_s[rt * 16 + m];
        pcc_s[m * 260 + ct0 * 16 + lm] = acc0[v] * rr * rn_s[ct0 * 16 + lm];
        pcc_s[m * 260 + ct1 * 16 + lm] = acc1[v] * rr * rn_s[ct1 * 16 + lm];
    }
    __syncthreads();

    if (wave < 4) {
        const int ot = wave;                   // output-channel tile (16 of 64)
        const _Float16* wrow = Wh + ((size_t)(ot * 16 + lm) * CC + c) * RR;
        v8f acc2 = {};
        for (int kb = 0; kb < RR; kb += 32) {
            v16h a2;
#pragma unroll
            for (int j = 0; j < 8; ++j) {
                a2[j]     = (_Float16)pcc_s[lm * 260 + kb + hi * 16 + j];
                a2[j + 8] = (_Float16)pcc_s[lm * 260 + kb + hi * 16 + 8 + j];
            }
            v16h bw = *(const v16h*)(wrow + kb + hi * 16);
            acc2 = wmma_f16(a2, bw, acc2);
        }
#pragma unroll
        for (int v = 0; v < 8; ++v) {
            const int m = v + hi * 8;
            atomicAdd(&e[((size_t)b * RR + rt * 16 + m) * C1 + ot * 16 + lm],
                      acc2[v]);
        }
    }
}

// ---------------------------------------------------------------------------
// Kernel 4: bias+leaky on e (in place), pooled mean over C1, 2-layer MLP,
// sigmoid attention -> d_out[4096 + b*256 + r]. One block per batch.
// ---------------------------------------------------------------------------
__global__ __launch_bounds__(256) void satt(float* __restrict__ e,
                                            const float* __restrict__ b_edge,
                                            const float* __restrict__ W1,
                                            const float* __restrict__ b1,
                                            const float* __restrict__ W2,
                                            const float* __restrict__ b2,
                                            float* __restrict__ out) {
    const int b = blockIdx.x, tid = threadIdx.x;
    __shared__ float pooled[256];
    __shared__ float hsh[64];
    pooled[tid] = 0.f;
    __syncthreads();

    float* eb = e + (size_t)b * RR * C1;
    for (int idx = tid; idx < RR * C1; idx += 256) {
        const int r = idx >> 6, o = idx & 63;
        float v = eb[idx] + b_edge[o];
        v = v > 0.f ? v : 0.01f * v;          // leaky_relu(0.01)
        eb[idx] = v;
        atomicAdd(&pooled[r], v);
    }
    __syncthreads();

    if (tid < 64) {
        float acc = b1[tid];
        for (int r = 0; r < RR; ++r)
            acc += W1[tid * RR + r] * (pooled[r] * (1.f / 64.f));
        hsh[tid] = acc > 0.f ? acc : 0.f;      // relu
    }
    __syncthreads();

    float acc = b2[tid];
    for (int j = 0; j < 64; ++j) acc += W2[tid * 64 + j] * hsh[j];
    out[BB * C2 + b * RR + tid] = 1.f / (1.f + __expf(-acc));
}

// ---------------------------------------------------------------------------
// Kernel 5: node conv: o[b,oc] = leaky(sum_{r,cc} att*e*W_node + b_node)
// One block per (b, oc); 256-thread reduction over 16384 terms.
// ---------------------------------------------------------------------------
__global__ __launch_bounds__(256) void nodeconv(const float* __restrict__ e,
                                                const float* __restrict__ outbuf,
                                                const float* __restrict__ W_node,
                                                const float* __restrict__ b_node,
                                                float* __restrict__ out) {
    const int b  = blockIdx.x >> 7;
    const int oc = blockIdx.x & 127;
    const int tid = threadIdx.x;
    const float* eb  = e + (size_t)b * RR * C1;
    const float* att = outbuf + BB * C2 + b * RR;

    float acc = 0.f;
    for (int idx = tid; idx < RR * C1; idx += 256) {
        const int r = idx & 255, cc = idx >> 8;
        acc += att[r] * eb[r * C1 + cc] * W_node[((size_t)oc * C1 + cc) * RR + r];
    }
    __shared__ float red[256];
    red[tid] = acc; __syncthreads();
    for (int s = 128; s > 0; s >>= 1) {
        if (tid < s) red[tid] += red[tid + s];
        __syncthreads();
    }
    if (tid == 0) {
        float v = red[0] + b_node[oc];
        out[b * C2 + oc] = v > 0.f ? v : 0.01f * v;
    }
}

// ---------------------------------------------------------------------------
extern "C" void kernel_launch(void* const* d_in, const int* in_sizes, int n_in,
                              void* d_out, int out_size, void* d_ws, size_t ws_size,
                              hipStream_t stream) {
    const float* x      = (const float*)d_in[0];
    const float* W_edge = (const float*)d_in[1];
    const float* b_edge = (const float*)d_in[2];
    const float* W1     = (const float*)d_in[3];
    const float* b1     = (const float*)d_in[4];
    const float* W2     = (const float*)d_in[5];
    const float* b2     = (const float*)d_in[6];
    const float* W_node = (const float*)d_in[7];
    const float* b_node = (const float*)d_in[8];
    float* out = (float*)d_out;

    char* ws = (char*)d_ws;
    _Float16* xc   = (_Float16*)(ws);                    // 134,217,728 B
    float*    rn   = (float*)   (ws + 134217728);        //     524,288 B
    _Float16* Wh   = (_Float16*)(ws + 134742016);        //     524,288 B
    float*    e    = (float*)   (ws + 135266304);        //   2,097,152 B

    hipMemsetAsync(e, 0, (size_t)BB * RR * C1 * sizeof(float), stream);

    prep_xc  <<<BB * RR,           256, 0, stream>>>(x, xc, rn);
    conv_wedge<<<(C1*CC*RR)/256,   256, 0, stream>>>(W_edge, Wh);
    corr_edge<<<BB * CC * (RR/16), 256, 0, stream>>>(xc, rn, Wh, e);
    satt     <<<BB,                256, 0, stream>>>(e, b_edge, W1, b1, W2, b2, out);
    nodeconv <<<BB * C2,           256, 0, stream>>>(e, out, W_node, b_node, out);
}
